// HierAgent_42700564857348
// MI455X (gfx1250) — matrix operand order repaired and checked by reading, not compiled, not run on hardware
//
#include <hip/hip_runtime.h>

#define N_AGENTS 16384
#define OBS 2048
#define HID 512
#define NACT 32
#define KSTEPS 64          // OBS / 32

typedef __attribute__((ext_vector_type(16))) __bf16 v16bf;
typedef __attribute__((ext_vector_type(8)))  float  v8f;
typedef __attribute__((ext_vector_type(4)))  unsigned int uint4v;

// ---------------- LDS layout (ushort units) ----------------
// A stage : 2 bufs x 64 rows x 40          [0, 5120)
// Wlds    : 48 rows x 520 (head weights)   [5120, 30080)
// P       : 6144 f32 partial head tiles    [30080, 42368)
// H       : 64 rows x 520 (hidden bf16)    [42368, 75648)
#define A_STRIDE 40
#define A_OFF(b) ((b) * 64 * A_STRIDE)
#define WL_OFF 5120
#define WL_STRIDE 520
#define P_OFF 30080
#define H_OFF 42368
#define H_STRIDE 520
#define LDS_USHORTS 75648   // 151296 bytes

__device__ __forceinline__ unsigned int f2bf(float x) {
  unsigned int u = __float_as_uint(x);
  return (u + 0x7fffu + ((u >> 16) & 1u)) >> 16;   // round-to-nearest-even
}

union Frag { v16bf v; uint4v q[2]; };

// --- pre-pass: W_enc f32 -> bf16, step-blocked layout wb[s][h][32] ---
__global__ __launch_bounds__(256)
void wenc_to_bf16(const float* __restrict__ w, unsigned short* __restrict__ o) {
  const int f  = blockIdx.x * 256 + threadIdx.x;  // float4-chunk id, 0..262143
  const int s  = f >> 12;                         // K-step (4096 chunks/step)
  const int r  = f & 4095;
  const int h  = r >> 3;
  const int c  = r & 7;
  const float4 v = *(const float4*)(w + (size_t)h * OBS + s * 32 + c * 4);
  const unsigned int p0 = f2bf(v.x) | (f2bf(v.y) << 16);
  const unsigned int p1 = f2bf(v.z) | (f2bf(v.w) << 16);
  *(unsigned long long*)(o + (size_t)s * 16384 + h * 32 + c * 4) =
      ((unsigned long long)p1 << 32) | p0;
}

__global__ __launch_bounds__(512, 1)
void hier_agent_fused(const float* __restrict__ x,
                      const unsigned short* __restrict__ wb,   // bf16 W_enc, blocked
                      const float* __restrict__ benc,
                      const float* __restrict__ Ws,
                      const float* __restrict__ bs,
                      const float* __restrict__ Wc,
                      const float* __restrict__ bc,
                      const int*   __restrict__ skill,
                      const int*   __restrict__ act,
                      float*       __restrict__ out)
{
  __shared__ unsigned short lds[LDS_USHORTS];

  const int tid  = threadIdx.x;
  const int blk  = blockIdx.x;
  const int lane = tid & 31;
  const int wave = tid >> 5;           // 0..15
  const int mt   = wave >> 2;          // 0..3  (16-row m-tile)
  const int nb   = (wave & 3) * 128;   // n-range base (8 n-tiles)
  const int half = lane >> 4;
  const int l15  = lane & 15;
  const int row0 = blk * 64;

  // A staging: 512 thr -> (row, float4-chunk of the 32-K slice)
  const int srow = tid >> 3;
  const int sc   = tid & 7;

  auto stageA = [&](int ks, int b) {
    const float4 v = *(const float4*)(x + (size_t)(row0 + srow) * OBS + ks + sc * 4);
    unsigned int* d = (unsigned int*)&lds[A_OFF(b) + srow * A_STRIDE + sc * 4];
    d[0] = f2bf(v.x) | (f2bf(v.y) << 16);
    d[1] = f2bf(v.z) | (f2bf(v.w) << 16);
  };

  // ---------------- GEMM: hidden = relu(x @ Wenc^T + benc) ----------------
  v8f acc[8];
#pragma unroll
  for (int t = 0; t < 8; ++t)
#pragma unroll
    for (int v = 0; v < 8; ++v) acc[t][v] = 0.f;

  const int aRowOff = (mt * 16 + l15) * A_STRIDE + half * 16;
  // per-lane base into blocked W (ushort units), step stride = 16384
  const size_t bBase = (size_t)(nb + l15) * 32 + half * 16;

  stageA(0, 0);
  __syncthreads();

  for (int s = 0; s < KSTEPS; ++s) {
    const int b = s & 1;

    // A frag from LDS (2x ds_load_b128)
    Frag fa;
    const uint4v* pa = (const uint4v*)&lds[A_OFF(b) + aRowOff];
    fa.q[0] = pa[0];
    fa.q[1] = pa[1];

    // 8 B frags straight from global (coalesced 1KB per wave per tile)
    Frag fb[8];
    const unsigned short* wstep = wb + (size_t)s * 16384 + bBase;
#pragma unroll
    for (int t = 0; t < 8; ++t) {
      const uint4v* pb = (const uint4v*)(wstep + t * 512);   // t*16 rows * 32
      fb[t].q[0] = pb[0];
      fb[t].q[1] = pb[1];
    }

    if (s + 1 < KSTEPS) stageA((s + 1) * 32, 1 - b);

#pragma unroll
    for (int t = 0; t < 8; ++t)
      acc[t] = __builtin_amdgcn_wmma_f32_16x16x32_bf16(
          false, fa.v, false, fb[t].v, (short)0, acc[t], false, false);

    __syncthreads();
  }

  // ---------------- bias + ReLU -> H (bf16, LDS) ----------------
#pragma unroll
  for (int t = 0; t < 8; ++t) {
    const int col = nb + t * 16 + l15;
    const float bv = benc[col];
#pragma unroll
    for (int v = 0; v < 8; ++v) {
      const int r = mt * 16 + v + half * 8;
      float h = acc[t][v] + bv;
      h = h > 0.f ? h : 0.f;
      lds[H_OFF + r * H_STRIDE + col] = (unsigned short)f2bf(h);
    }
  }

  // head weights bf16 -> Wlds[48][520]: rows 0..31 Ws, 32 Wc, 33..47 zero
#pragma unroll
  for (int n = 0; n < 48; ++n) {
    float wv = 0.f;
    if (n < NACT)       wv = Ws[n * HID + tid];
    else if (n == NACT) wv = Wc[tid];
    lds[WL_OFF + n * WL_STRIDE + tid] = (unsigned short)f2bf(wv);
  }
  __syncthreads();

  // ---------------- heads via WMMA: [64x512] @ [512x48] ----------------
  float* Pf = (float*)&lds[P_OFF];
  if (wave < 8) {
    const int m = wave & 3;          // m-tile
    const int g = wave >> 2;         // K half (8 steps each)
    v8f hacc[3];
#pragma unroll
    for (int nt = 0; nt < 3; ++nt)
#pragma unroll
      for (int v = 0; v < 8; ++v) hacc[nt][v] = 0.f;

    const int haRow = (m * 16 + l15) * H_STRIDE;
#pragma unroll
    for (int ks = 0; ks < 8; ++ks) {
      const int kb = (g * 8 + ks) * 32;
      Frag fa;
      const uint4v* pa = (const uint4v*)&lds[H_OFF + haRow + kb + half * 16];
      fa.q[0] = pa[0];
      fa.q[1] = pa[1];
#pragma unroll
      for (int nt = 0; nt < 3; ++nt) {
        Frag fbh;
        const uint4v* pw =
            (const uint4v*)&lds[WL_OFF + (nt * 16 + l15) * WL_STRIDE + kb + half * 16];
        fbh.q[0] = pw[0];
        fbh.q[1] = pw[1];
        hacc[nt] = __builtin_amdgcn_wmma_f32_16x16x32_bf16(
            false, fa.v, false, fbh.v, (short)0, hacc[nt], false, false);
      }
    }
    // store partial tiles: tile id = ((m*2+g)*3 + nt)
#pragma unroll
    for (int nt = 0; nt < 3; ++nt) {
      const int tix = ((m * 2 + g) * 3 + nt) * 256;
#pragma unroll
      for (int v = 0; v < 8; ++v)
        Pf[tix + (v + half * 8) * 16 + l15] = hacc[nt][v];
    }
  }
  __syncthreads();

  // ---------------- finalize: one thread per row ----------------
  if (tid < 64) {
    const int r  = tid;
    const int m  = r >> 4;
    const int mr = r & 15;
    const int gr = row0 + r;
    const int an = act[gr];

    float logits[NACT];
    float mx = -3.0e38f;
#pragma unroll
    for (int a = 0; a < NACT; ++a) {
      const int nt = a >> 4, nc = a & 15;
      const float pv = Pf[((m * 2 + 0) * 3 + nt) * 256 + mr * 16 + nc] +
                       Pf[((m * 2 + 1) * 3 + nt) * 256 + mr * 16 + nc];
      logits[a] = pv + bs[a];
      mx = fmaxf(mx, logits[a]);
    }
    const float val = Pf[((m * 2 + 0) * 3 + 2) * 256 + mr * 16] +
                      Pf[((m * 2 + 1) * 3 + 2) * 256 + mr * 16] + bc[0];

    float Z = 0.f, S1 = 0.f, la = 0.f;
#pragma unroll
    for (int a = 0; a < NACT; ++a) {
      const float e = __expf(logits[a] - mx);
      Z += e;
      S1 += e * logits[a];
      la = (a == an) ? logits[a] : la;
    }
    const float logZ = __logf(Z);

    out[0 * N_AGENTS + gr] = (float)an;
    out[1 * N_AGENTS + gr] = (float)skill[gr];
    out[2 * N_AGENTS + gr] = -2.7725887f;          // -log(16)
    out[3 * N_AGENTS + gr] = la - mx - logZ;       // skill_logprob
    out[4 * N_AGENTS + gr] = (mx + logZ) - S1 / Z; // entropy
    out[5 * N_AGENTS + gr] = val;                  // value
  }
}

extern "C" void kernel_launch(void* const* d_in, const int* in_sizes, int n_in,
                              void* d_out, int out_size, void* d_ws, size_t ws_size,
                              hipStream_t stream) {
  const float* x    = (const float*)d_in[0];
  const float* Wenc = (const float*)d_in[1];
  const float* benc = (const float*)d_in[2];
  // d_in[3] = Wm, d_in[4] = bm: dead w.r.t. reference outputs
  const float* Ws   = (const float*)d_in[5];
  const float* bs   = (const float*)d_in[6];
  const float* Wc   = (const float*)d_in[7];
  const float* bc   = (const float*)d_in[8];
  const int* skill  = (const int*)d_in[9];
  const int* act    = (const int*)d_in[10];
  float* out = (float*)d_out;

  unsigned short* wb = (unsigned short*)d_ws;      // 2 MB bf16 blocked W_enc

  hipLaunchKernelGGL(wenc_to_bf16, dim3(1024), dim3(256), 0, stream, Wenc, wb);
  hipLaunchKernelGGL(hier_agent_fused, dim3(N_AGENTS / 64), dim3(512), 0, stream,
                     x, wb, benc, Ws, bs, Wc, bc, skill, act, out);
}
